// TemporalMultiheadAttention_61589831024980
// MI455X (gfx1250) — compile-verified, matrix operand order checked by reading, not run on hardware
//
#include <hip/hip_runtime.h>
#include <stdint.h>

// ---------------------------------------------------------------------------
// Temporal multi-head attention for MI455X (gfx1250, wave32, WMMA).
//   B=2, T=64, H=W=32, C=128, heads=4, d=32.  scale = 1/sqrt(T) = 1/8.
// Pipeline: bf16 weights (transposed) -> fused QKV projection (WMMA bf16)
//           -> per-group attention (TDM load to LDS + WMMA + softmax)
//           -> output projection (WMMA bf16, f32 out).
// ---------------------------------------------------------------------------

typedef __bf16 bf16;
typedef __attribute__((ext_vector_type(16))) __bf16    v16bf;
typedef __attribute__((ext_vector_type(8)))  float     v8f;
typedef __attribute__((ext_vector_type(4)))  uint32_t  u32x4;
typedef __attribute__((ext_vector_type(8)))  int       i32x8;
typedef __attribute__((ext_vector_type(4)))  int       i32x4;

#define B_   2
#define T_   64
#define H_   32
#define W_   32
#define C_   128
#define NH_  4
#define DH_  32
#define TOK_ (B_*T_*H_*W_)    // 131072 tokens
#define NG_  (B_*H_*W_*NH_)   // 8192 attention groups
#define GSTRIDE_ 6144         // per-group bf16 elems: K(2048) V(2048) Q(2048)

// K-index inside a 16x32 bf16 A/B operand for element e of the per-lane v16bf,
// per CDNA5 ISA 7.12.2 (16-bit A-matrix layout; B mirrors with N<->M roles).
__device__ __forceinline__ int kmap(int e, int khalf) {
  return ((e >> 3) << 4) + (khalf << 3) + (e & 7);
}

__device__ __forceinline__ v8f zero8() {
  v8f z;
#pragma unroll
  for (int i = 0; i < 8; ++i) z[i] = 0.0f;
  return z;
}

__device__ __forceinline__ v8f wmma_bf16(v16bf a, v16bf b, v8f c) {
  return __builtin_amdgcn_wmma_f32_16x16x32_bf16(false, a, false, b,
                                                 (short)0, c, false, false);
}

// ---------------------------------------------------------------------------
// Kernel 1: convert + transpose weights to bf16 so WMMA B-operand gathers are
// row-contiguous (two 16-byte chunks per lane).
// ---------------------------------------------------------------------------
__global__ void __launch_bounds__(256)
cvt_weights(const float* __restrict__ Wkv, const float* __restrict__ Wq,
            const float* __restrict__ Wp,
            bf16* __restrict__ wkvT, bf16* __restrict__ wqT,
            bf16* __restrict__ wpT) {
  int i = blockIdx.x * 256 + threadIdx.x;
  if (i < 256 * 128) {                     // WkvT[o][c] = Wkv[c][o], o<256
    int o = i >> 7, c = i & 127;
    wkvT[i] = (bf16)Wkv[c * 256 + o];
  }
  if (i < 128 * 128) {                     // WqT / WpT, o<128
    int o = i >> 7, c = i & 127;
    wqT[i] = (bf16)Wq[c * 128 + o];
    wpT[i] = (bf16)Wp[c * 128 + o];
  }
}

// ---------------------------------------------------------------------------
// Kernel 2: fused KV + Q projection.
// Block = 64 tokens (4 waves x M=16). A tiles (4 k-steps) kept in VGPRs and
// reused across every N-tile. Output scattered into grouped [g][K|V|Q][t][d]
// bf16 layout so the attention kernel gets one contiguous 12 KB block.
// ---------------------------------------------------------------------------
__global__ void __launch_bounds__(128)
proj_qkv(const float* __restrict__ X, const float* __restrict__ Y,
         const bf16* __restrict__ wkvT, const float* __restrict__ bkv,
         const bf16* __restrict__ wqT,  const float* __restrict__ bq,
         bf16* __restrict__ kvq) {
  const int lane  = threadIdx.x & 31;
  const int wv    = threadIdx.x >> 5;
  const int khalf = lane >> 4;
  const int nlo   = lane & 15;
  const int tok0  = blockIdx.x * 64 + wv * 16;

  // A operands: this lane supplies row (tok0 + nlo), k-half per lane group.
  const int arow = tok0 + nlo;
  v16bf aX[4], aY[4];
#pragma unroll
  for (int ks = 0; ks < 4; ++ks) {
    const float* xr = X + (size_t)arow * C_ + ks * 32;
    const float* yr = Y + (size_t)arow * C_ + ks * 32;
#pragma unroll
    for (int e = 0; e < 16; ++e) {
      int k = kmap(e, khalf);
      aX[ks][e] = (bf16)xr[k];
      aY[ks][e] = (bf16)yr[k];
    }
  }

  // ---- KV projection: N = 256 -> 16 N-tiles --------------------------------
  for (int nt = 0; nt < 16; ++nt) {
    const int nOut = nt * 16 + nlo;
    v8f acc;
    {
      float bias = bkv[nOut];
#pragma unroll
      for (int r = 0; r < 8; ++r) acc[r] = bias;
    }
#pragma unroll
    for (int ks = 0; ks < 4; ++ks) {
      v16bf bm;
      const bf16* br = wkvT + nOut * C_ + ks * 32;
#pragma unroll
      for (int e = 0; e < 16; ++e) bm[e] = br[kmap(e, khalf)];
      acc = wmma_bf16(aX[ks], bm, acc);
    }
    // KV[..., n, d, 2]: nOut = nh*64 + dd*2 + sel
    const int nh  = nOut >> 6;
    const int rem = nOut & 63;
    const int dd  = rem >> 1;
    const int sel = rem & 1;                 // 0 -> K slot, 1 -> V slot
#pragma unroll
    for (int r = 0; r < 8; ++r) {
      int token = tok0 + 8 * khalf + r;      // D-layout row for this lane
      int wsp = token & 31, hsp = (token >> 5) & 31;
      int t = (token >> 10) & 63, bb = token >> 16;
      int g = ((bb * H_ + hsp) * W_ + wsp) * NH_ + nh;
      kvq[(size_t)g * GSTRIDE_ + sel * 2048 + t * DH_ + dd] = (bf16)acc[r];
    }
  }

  // ---- Q projection: N = 128 -> 8 N-tiles ----------------------------------
  for (int nt = 0; nt < 8; ++nt) {
    const int nOut = nt * 16 + nlo;
    v8f acc;
    {
      float bias = bq[nOut];
#pragma unroll
      for (int r = 0; r < 8; ++r) acc[r] = bias;
    }
#pragma unroll
    for (int ks = 0; ks < 4; ++ks) {
      v16bf bm;
      const bf16* br = wqT + nOut * C_ + ks * 32;
#pragma unroll
      for (int e = 0; e < 16; ++e) bm[e] = br[kmap(e, khalf)];
      acc = wmma_bf16(aY[ks], bm, acc);
    }
    const int nh = nOut >> 5;
    const int dd = nOut & 31;
#pragma unroll
    for (int r = 0; r < 8; ++r) {
      int token = tok0 + 8 * khalf + r;
      int wsp = token & 31, hsp = (token >> 5) & 31;
      int t = (token >> 10) & 63, bb = token >> 16;
      int g = ((bb * H_ + hsp) * W_ + wsp) * NH_ + nh;
      kvq[(size_t)g * GSTRIDE_ + 4096 + t * DH_ + dd] = (bf16)acc[r];
    }
  }
}

// ---------------------------------------------------------------------------
// Kernel 3: attention for one (b,h,w,head) group.  T=64 queries/keys, d=32.
// K/V/Q (12 KB contiguous) pulled into LDS by the Tensor Data Mover, then:
//   S = Q K^T (4 WMMAs/wave, K-dim=32 exact), softmax over rows (shfl-xor
//   within the 16-lane half that owns each row), P->LDS bf16, O = P V.
// ---------------------------------------------------------------------------
__global__ void __launch_bounds__(128)
attn(const bf16* __restrict__ kvq, bf16* __restrict__ attnout) {
  __shared__ bf16 shKVQ[GSTRIDE_];   // K[64][32] | V[64][32] | Q[64][32]
  __shared__ bf16 shP[64 * 64];      // softmax probabilities, row-major

  const int g     = blockIdx.x;
  const int lane  = threadIdx.x & 31;
  const int wv    = threadIdx.x >> 5;
  const int khalf = lane >> 4;
  const int nlo   = lane & 15;
  const int nh  = g & 3;
  const int wsp = (g >> 2) & 31;
  const int hsp = (g >> 7) & 31;
  const int bb  = g >> 12;

  // --- one TDM copy of the whole group's K/V/Q block into LDS (wave 0) -----
  if (threadIdx.x < 32) {
#if __has_builtin(__builtin_amdgcn_tensor_load_to_lds)
    uint64_t ga   = (uint64_t)(uintptr_t)(kvq + (size_t)g * GSTRIDE_);
    uint32_t ldsa = (uint32_t)(uintptr_t)(&shKVQ[0]);
    // D# group 0: count=1 | lds_addr | global_addr[56:0] | type=2
    u32x4 g0 = { 1u, ldsa, (uint32_t)ga,
                 (uint32_t)((ga >> 32) & 0x1FFFFFFu) | (2u << 30) };
    // D# group 1: data_size=2B, 1-D tile: tensor_dim0=tile_dim0=6144 elems
    i32x8 g1 = { (int)(1u << 16),        // workgroup_mask=0, data_size=1 (2B)
                 (int)(6144u << 16),     // tensor_dim0[15:0] -> bits[63:48]
                 (int)(1u << 16),        // tensor_dim0 hi=0, tensor_dim1=1
                 (int)(6144u << 16),     // tile_dim0=6144 -> bits[127:112]
                 0,                      // tile_dim1=0, tile_dim2=0 (unused)
                 6144,                   // tensor_dim0_stride lo
                 0, 0 };
    i32x4 z4 = { 0, 0, 0, 0 };
    i32x8 z8 = { 0, 0, 0, 0, 0, 0, 0, 0 };
    __builtin_amdgcn_tensor_load_to_lds(g0, g1, z4, z4, z8, 0);
    __builtin_amdgcn_s_wait_tensorcnt(0);
#else
    for (int i = lane; i < GSTRIDE_; i += 32)
      shKVQ[i] = kvq[(size_t)g * GSTRIDE_ + i];
#endif
  }
  __syncthreads();

  const bf16* shK = shKVQ;
  const bf16* shV = shKVQ + 2048;
  const bf16* shQ = shKVQ + 4096;

  // ---- S = Q K^T : wave wv owns query rows [16*wv, 16*wv+16) ---------------
  v16bf aq;
#pragma unroll
  for (int e = 0; e < 16; ++e)
    aq[e] = shQ[(wv * 16 + nlo) * DH_ + kmap(e, khalf)];

  v8f s[4];
#pragma unroll
  for (int nt = 0; nt < 4; ++nt) {
    v16bf bk;                                       // B[k][j] = K[j][k]
#pragma unroll
    for (int e = 0; e < 16; ++e)
      bk[e] = shK[(nt * 16 + nlo) * DH_ + kmap(e, khalf)];
    s[nt] = wmma_bf16(aq, bk, zero8());
  }

  // ---- softmax over each row (64 values spread over 16 lanes x 4 tiles) ----
  const float scale = 0.125f;                        // 1/sqrt(T)
  float p[4][8];
#pragma unroll
  for (int r = 0; r < 8; ++r) {
    float mx = -3.0e38f;
#pragma unroll
    for (int nt = 0; nt < 4; ++nt) {
      p[nt][r] = s[nt][r] * scale;
      mx = fmaxf(mx, p[nt][r]);
    }
#pragma unroll
    for (int m = 1; m <= 8; m <<= 1) mx = fmaxf(mx, __shfl_xor(mx, m, 32));
    float sum = 0.0f;
#pragma unroll
    for (int nt = 0; nt < 4; ++nt) {
      p[nt][r] = __expf(p[nt][r] - mx);
      sum += p[nt][r];
    }
#pragma unroll
    for (int m = 1; m <= 8; m <<= 1) sum += __shfl_xor(sum, m, 32);
    float inv = 1.0f / sum;
#pragma unroll
    for (int nt = 0; nt < 4; ++nt)
      shP[(wv * 16 + 8 * khalf + r) * 64 + nt * 16 + nlo] =
          (bf16)(p[nt][r] * inv);
  }
  __syncthreads();

  // ---- O = P V : M=16 rows/wave, K=64 (2 k-steps), N=32 (2 n-tiles) --------
  v16bf ap[2];
#pragma unroll
  for (int kk = 0; kk < 2; ++kk)
#pragma unroll
    for (int e = 0; e < 16; ++e)
      ap[kk][e] = shP[(wv * 16 + nlo) * 64 + kk * 32 + kmap(e, khalf)];

#pragma unroll
  for (int nt = 0; nt < 2; ++nt) {
    v8f acc = zero8();
#pragma unroll
    for (int kk = 0; kk < 2; ++kk) {
      v16bf bv;                                     // B[k][n] = V[k][n]
#pragma unroll
      for (int e = 0; e < 16; ++e)
        bv[e] = shV[(kk * 32 + kmap(e, khalf)) * DH_ + nt * 16 + nlo];
      acc = wmma_bf16(ap[kk], bv, acc);
    }
    const int dd = nt * 16 + nlo;
    const int ch = nh * DH_ + dd;
#pragma unroll
    for (int r = 0; r < 8; ++r) {
      int t = wv * 16 + 8 * khalf + r;
      int token = ((bb * T_ + t) * H_ + hsp) * W_ + wsp;
      attnout[(size_t)token * C_ + ch] = (bf16)acc[r];
    }
  }
}

// ---------------------------------------------------------------------------
// Kernel 4: output projection, f32 result + bias.
// ---------------------------------------------------------------------------
__global__ void __launch_bounds__(128)
proj_out(const bf16* __restrict__ attnout, const bf16* __restrict__ wpT,
         const float* __restrict__ bp, float* __restrict__ out) {
  const int lane  = threadIdx.x & 31;
  const int wv    = threadIdx.x >> 5;
  const int khalf = lane >> 4;
  const int nlo   = lane & 15;
  const int tok0  = blockIdx.x * 64 + wv * 16;
  const int arow  = tok0 + nlo;

  v16bf a[4];
#pragma unroll
  for (int ks = 0; ks < 4; ++ks) {
    const bf16* xr = attnout + (size_t)arow * C_ + ks * 32;
#pragma unroll
    for (int e = 0; e < 16; ++e) a[ks][e] = xr[kmap(e, khalf)];
  }

  for (int nt = 0; nt < 8; ++nt) {
    const int nOut = nt * 16 + nlo;
    v8f acc;
    {
      float bias = bp[nOut];
#pragma unroll
      for (int r = 0; r < 8; ++r) acc[r] = bias;
    }
#pragma unroll
    for (int ks = 0; ks < 4; ++ks) {
      v16bf bm;
      const bf16* br = wpT + nOut * C_ + ks * 32;
#pragma unroll
      for (int e = 0; e < 16; ++e) bm[e] = br[kmap(e, khalf)];
      acc = wmma_bf16(a[ks], bm, acc);
    }
#pragma unroll
    for (int r = 0; r < 8; ++r) {
      int token = tok0 + 8 * khalf + r;
      out[(size_t)token * C_ + nOut] = acc[r];
    }
  }
}

// ---------------------------------------------------------------------------
// Host launcher.
// ---------------------------------------------------------------------------
extern "C" void kernel_launch(void* const* d_in, const int* in_sizes, int n_in,
                              void* d_out, int out_size, void* d_ws,
                              size_t ws_size, hipStream_t stream) {
  const float* X   = (const float*)d_in[0];
  const float* Y   = (const float*)d_in[1];
  const float* Wkv = (const float*)d_in[2];
  const float* bkv = (const float*)d_in[3];
  const float* Wq  = (const float*)d_in[4];
  const float* bq  = (const float*)d_in[5];
  const float* Wp  = (const float*)d_in[6];
  const float* bp  = (const float*)d_in[7];
  float* out = (float*)d_out;

  // Workspace layout (bytes):
  char* ws = (char*)d_ws;
  bf16* wkvT = (bf16*)(ws + 0);                       //  64 KB
  bf16* wqT  = (bf16*)(ws + 65536);                   //  32 KB
  bf16* wpT  = (bf16*)(ws + 98304);                   //  32 KB
  bf16* kvq  = (bf16*)(ws + 131072);                  //  96 MB grouped K/V/Q
  bf16* aout = (bf16*)(ws + 131072 + (size_t)NG_ * GSTRIDE_ * 2);  // 32 MB

  cvt_weights<<<128, 256, 0, stream>>>(Wkv, Wq, Wp, wkvT, wqT, wpT);
  proj_qkv<<<TOK_ / 64, 128, 0, stream>>>(X, Y, wkvT, bkv, wqT, bq, kvq);
  attn<<<NG_, 128, 0, stream>>>(kvq, aout);
  proj_out<<<TOK_ / 64, 128, 0, stream>>>(aout, wpT, bp, out);
}
